// AttentionBasicBlock_74964359184915
// MI455X (gfx1250) — matrix operand order repaired and checked by reading, not compiled, not run on hardware
//
#include <hip/hip_runtime.h>

typedef __attribute__((ext_vector_type(16))) __bf16 v16bf;
typedef __attribute__((ext_vector_type(8)))  __bf16 v8bf;
typedef __attribute__((ext_vector_type(8)))  float  v8f;

#define B_   32
#define C_   48
#define N_   2048
#define NA_  512
#define CQ_  4
#define CS_  3
#define KW_  32

// ---------------- workspace layout (float offsets) ----------------
constexpr int OFF_AV     = 0;                       // B*C
constexpr int OFF_SQK    = 1536;                    // B*CQ
constexpr int OFF_SQQ    = 1664;
constexpr int OFF_SQA    = 1792;
constexpr int OFF_INVK   = 1920;
constexpr int OFF_INVQ   = 2048;
constexpr int OFF_INVA   = 2176;
constexpr int OFF_BN1SUM = 2304;                    // C
constexpr int OFF_BN1SQ  = 2352;
constexpr int OFF_BN1MU  = 2400;
constexpr int OFF_BN1RS  = 2448;
constexpr int OFF_BN2SUM = 2496;
constexpr int OFF_BN2SQ  = 2544;
constexpr int OFF_BN2MU  = 2592;
constexpr int OFF_BN2RS  = 2640;
constexpr int OFF_PK     = 4096;                    // B*CQ*N
constexpr int OFF_PQ     = OFF_PK + B_*CQ_*N_;
constexpr int OFF_ANC    = OFF_PQ + B_*CQ_*N_;      // B*CQ*NA
constexpr int OFF_PV     = OFF_ANC + B_*CQ_*NA_;    // B*C*N
constexpr int OFF_ATTN   = OFF_PV + B_*C_*N_;
constexpr int OFF_DWC    = OFF_ATTN + B_*C_*N_;

constexpr int SMEM_PROJ = (C_*256 + C_*C_ + 3*CQ_*C_ + CQ_*256 + 16) * 4;          // ~64.8 KB
constexpr int SMEM_ATTN = 8192 + 2048 + 2048 + C_*NA_*2 + (CQ_*128*4 + NA_*128*2); // 194560
constexpr int SMEM_DWC  = (C_*288 + 2*C_) * 4;                                     // ~55.7 KB

// ---------------- device helpers ----------------
__device__ inline float wave_sum(float v) {
  #pragma unroll
  for (int o = 16; o > 0; o >>= 1) v += __shfl_xor(v, o);
  return v;
}
__device__ inline float wave_max(float v) {
  #pragma unroll
  for (int o = 16; o > 0; o >>= 1) v = fmaxf(v, __shfl_xor(v, o));
  return v;
}
__device__ inline float eluf(float v) { return v > 0.f ? v : expf(v) - 1.f; }

// A fragment (16x32 bf16) from row-major LDS (stride ld elements).
// CDNA5 layout: lane m=lane&15, kb=(lane<16)?0:8; elems 0..7 -> K=kb..kb+7,
// elems 8..15 -> K=kb+16..kb+23  => two contiguous 16B runs -> 2x ds_load_b128.
__device__ inline v16bf load_a(const __bf16* src, int ld) {
  int lane = threadIdx.x & 31;
  int m  = lane & 15;
  int kb = (lane < 16) ? 0 : 8;
  v8bf lo = *(const v8bf*)(src + m * ld + kb);
  v8bf hi = *(const v8bf*)(src + m * ld + kb + 16);
  return __builtin_shufflevector(lo, hi, 0, 1, 2, 3, 4, 5, 6, 7,
                                 8, 9, 10, 11, 12, 13, 14, 15);
}
// B fragment (32x16 bf16) from TRANSPOSED LDS storage BT[N][K] (stride ld):
// lane n=lane&15, kb=(lane<16)?0:8; elems 0..7 -> K=kb..kb+7,
// elems 8..15 -> K=kb+16..kb+23 -> 2x ds_load_b128.
__device__ inline v16bf load_bt(const __bf16* src, int ld) {
  int lane = threadIdx.x & 31;
  int n  = lane & 15;
  int kb = (lane < 16) ? 0 : 8;
  v8bf lo = *(const v8bf*)(src + n * ld + kb);
  v8bf hi = *(const v8bf*)(src + n * ld + kb + 16);
  return __builtin_shufflevector(lo, hi, 0, 1, 2, 3, 4, 5, 6, 7,
                                 8, 9, 10, 11, 12, 13, 14, 15);
}

// ---------------- kernels ----------------
__global__ void k_zero(float* ws) {
  for (int i = threadIdx.x; i < 2688 - 1536; i += 256) ws[1536 + i] = 0.f;
}

__global__ void k_se(const float* __restrict__ x, const float* __restrict__ w1,
                     const float* __restrict__ b1, const float* __restrict__ w2,
                     const float* __restrict__ b2, float* __restrict__ av) {
  __shared__ float means[C_];
  __shared__ float hid[CS_];
  int b = blockIdx.x;
  int lane = threadIdx.x & 31, wave = threadIdx.x >> 5;
  for (int c = wave; c < C_; c += 8) {
    const float* xp = x + ((size_t)b * C_ + c) * N_;
    float s = 0.f;
    for (int n = lane; n < N_; n += 32) s += xp[n];
    s = wave_sum(s);
    if (lane == 0) means[c] = s * (1.f / N_);
  }
  __syncthreads();
  if (threadIdx.x < CS_) {
    float a = b1[threadIdx.x];
    #pragma unroll 8
    for (int c = 0; c < C_; ++c) a += w1[threadIdx.x * C_ + c] * means[c];
    hid[threadIdx.x] = eluf(a);
  }
  __syncthreads();
  if (threadIdx.x < C_) {
    float g = b2[threadIdx.x];
    #pragma unroll
    for (int s = 0; s < CS_; ++s) g += w2[threadIdx.x * CS_ + s] * hid[s];
    av[b * C_ + threadIdx.x] = 1.f / (1.f + expf(-g));
  }
}

__global__ void __launch_bounds__(256) k_proj(
    const float* __restrict__ x, const float* __restrict__ av,
    const float* __restrict__ kw, const float* __restrict__ kb,
    const float* __restrict__ qw, const float* __restrict__ qb,
    const float* __restrict__ vw, const float* __restrict__ vb,
    const float* __restrict__ aw, const float* __restrict__ ab,
    float* __restrict__ pk, float* __restrict__ pq, float* __restrict__ pv,
    float* __restrict__ anc, float* sqk, float* sqq, float* sqa) {
  extern __shared__ __align__(16) char smem[];
  float* ar   = (float*)smem;            // 48*256
  float* wv_s = ar + C_ * 256;           // 48*48
  float* wk_s = wv_s + C_ * C_;          // 4*48
  float* wq_s = wk_s + CQ_ * C_;
  float* wa_s = wq_s + CQ_ * C_;
  float* pa_s = wa_s + CQ_ * C_;         // 4*256
  float* ssq  = pa_s + CQ_ * 256;        // 12
  int b = blockIdx.y, n0 = blockIdx.x * 256, tid = threadIdx.x;
  if (tid < 12) ssq[tid] = 0.f;
  for (int i = tid; i < C_ * C_; i += 256) wv_s[i] = vw[i];
  for (int i = tid; i < CQ_ * C_; i += 256) { wk_s[i] = kw[i]; wq_s[i] = qw[i]; wa_s[i] = aw[i]; }
  for (int i = tid; i < C_ * 256; i += 256) {
    int c = i >> 8, j = i & 255;
    ar[i] = x[((size_t)b * C_ + c) * N_ + n0 + j] * (1.f + av[b * C_ + c]);
  }
  __syncthreads();
  for (int i = tid; i < C_ * 256; i += 256) {          // value projection
    int o = i >> 8, j = i & 255;
    float acc = vb[o];
    #pragma unroll 8
    for (int c = 0; c < C_; ++c) acc += wv_s[o * C_ + c] * ar[c * 256 + j];
    pv[((size_t)b * C_ + o) * N_ + n0 + j] = acc;
  }
  for (int i = tid; i < CQ_ * 256; i += 256) {         // k / q / anchor projections
    int o = i >> 8, j = i & 255;
    float ak = kb[o], aq = qb[o], aa = ab[o];
    #pragma unroll 8
    for (int c = 0; c < C_; ++c) {
      float v = ar[c * 256 + j];
      ak += wk_s[o * C_ + c] * v; aq += wq_s[o * C_ + c] * v; aa += wa_s[o * C_ + c] * v;
    }
    pk[((size_t)b * CQ_ + o) * N_ + n0 + j] = ak;
    pq[((size_t)b * CQ_ + o) * N_ + n0 + j] = aq;
    pa_s[o * 256 + j] = aa;
    atomicAdd(&ssq[o], ak * ak);
    atomicAdd(&ssq[4 + o], aq * aq);
  }
  __syncthreads();
  for (int i = tid; i < CQ_ * 64; i += 256) {          // avg-pool anchor by 4
    int o = i >> 6, aj = i & 63;
    float m = 0.25f * (pa_s[o * 256 + 4 * aj] + pa_s[o * 256 + 4 * aj + 1] +
                       pa_s[o * 256 + 4 * aj + 2] + pa_s[o * 256 + 4 * aj + 3]);
    anc[((size_t)b * CQ_ + o) * NA_ + (n0 >> 2) + aj] = m;
    atomicAdd(&ssq[8 + o], m * m);
  }
  __syncthreads();
  if (tid < CQ_) {
    atomicAdd(&sqk[b * CQ_ + tid], ssq[tid]);
    atomicAdd(&sqq[b * CQ_ + tid], ssq[4 + tid]);
    atomicAdd(&sqa[b * CQ_ + tid], ssq[8 + tid]);
  }
}

__global__ void k_norms(float* ws) {
  int i = threadIdx.x;
  if (i < B_ * CQ_) {
    ws[OFF_INVK + i] = 1.f / fmaxf(sqrtf(ws[OFF_SQK + i]), 1e-12f);
    ws[OFF_INVQ + i] = 1.f / fmaxf(sqrtf(ws[OFF_SQQ + i]), 1e-12f);
    ws[OFF_INVA + i] = 1.f / fmaxf(sqrtf(ws[OFF_SQA + i]), 1e-12f);
  }
}

__global__ void __launch_bounds__(256) k_attention(
    const float* __restrict__ pkg, const float* __restrict__ pqg,
    const float* __restrict__ ancg, const float* __restrict__ pvg,
    const float* __restrict__ invk, const float* __restrict__ invq,
    const float* __restrict__ inva, float* __restrict__ attn) {
  extern __shared__ __align__(16) char smem[];
  float*  anch4 = (float*)smem;                 // [NA][4] f32, float4 per anchor
  float*  mq    = anch4 + CQ_ * NA_;            // 512
  float*  sqv   = mq + NA_;                     // 512
  __bf16* zds   = (__bf16*)(sqv + NA_);         // 48*512 bf16 (row-major [M][K])
  char*   region = (char*)(zds + C_ * NA_);     // overlapped phase scratch
  // phase 0 view
  float*  pqf4 = (float*)region;                // [N][4] f32
  // phase 1 view
  float*  pkc4 = (float*)region;                // [64][4] f32
  __bf16* PkT  = (__bf16*)(pkc4 + CQ_ * 64);    // transposed probs [a=512][n=64]
  __bf16* pvbf = PkT + NA_ * 64;                // [48][64]
  // phase 3 view
  float*  pqc4 = (float*)region;                // [128][4] f32
  __bf16* PQT  = (__bf16*)(pqc4 + CQ_ * 128);   // transposed probs [n=128][a=512]

  int b = blockIdx.x, tid = threadIdx.x;
  int lane = tid & 31, wave = tid >> 5;

  // ---- phase 0: anchor_n to LDS (channel-interleaved); pq softmax stats ----
  for (int i = tid; i < CQ_ * NA_; i += 256) {
    int c = i & 3, a = i >> 2;
    anch4[i] = ancg[((size_t)b * CQ_ + c) * NA_ + a] * inva[b * CQ_ + c];
  }
  for (int i = tid; i < CQ_ * N_; i += 256) {
    int c = i & 3, n = i >> 2;
    pqf4[i] = pqg[((size_t)b * CQ_ + c) * N_ + n] * invq[b * CQ_ + c];
  }
  __syncthreads();
  for (int a = tid; a < NA_; a += 256) {
    float4 av4 = *(const float4*)(anch4 + 4 * a);
    float m = -3.4e38f;
    for (int n = 0; n < N_; ++n) {
      float4 q = *(const float4*)(pqf4 + 4 * n);
      float e = av4.x * q.x + av4.y * q.y + av4.z * q.z + av4.w * q.w;
      m = fmaxf(m, e);
    }
    float s = 0.f;
    for (int n = 0; n < N_; ++n) {
      float4 q = *(const float4*)(pqf4 + 4 * n);
      float e = av4.x * q.x + av4.y * q.y + av4.z * q.z + av4.w * q.w;
      s += expf(e - m);
    }
    mq[a] = m; sqv[a] = 1.f / s;
  }
  __syncthreads();

  // ---- phase 1: z[48,512] += pv[48,64] x softmax(pk_e)[64,512], chunked ----
  v8f zero8 = {0.f, 0.f, 0.f, 0.f, 0.f, 0.f, 0.f, 0.f};
  v8f accz[3][4];
  #pragma unroll
  for (int mt = 0; mt < 3; ++mt)
    #pragma unroll
    for (int j = 0; j < 4; ++j) accz[mt][j] = zero8;

  for (int ch = 0; ch < N_ / 64; ++ch) {
    int nc = ch * 64;
    // prefetch next chunk of pv (global_prefetch_b8 path)
    if (ch + 1 < N_ / 64) {
      const float* nxt = pvg + (size_t)b * C_ * N_ + nc + 64;
      for (int i = tid; i < C_; i += 256) __builtin_prefetch(nxt + (size_t)i * N_, 0, 1);
    }
    for (int i = tid; i < CQ_ * 64; i += 256) {
      int c = i & 3, n = i >> 2;
      pkc4[i] = pkg[((size_t)b * CQ_ + c) * N_ + nc + n] * invk[b * CQ_ + c];
    }
    for (int i = tid; i < C_ * 64; i += 256) {
      int c = i >> 6;
      pvbf[i] = (__bf16)pvg[((size_t)b * C_ + c) * N_ + nc + (i & 63)];
    }
    __syncthreads();
    // softmax over 512 anchors; 8 rows per wave; write transposed PkT[a][n]
    #pragma unroll 1
    for (int rr = 0; rr < 8; ++rr) {
      int r = wave * 8 + rr;
      float4 kv = *(const float4*)(pkc4 + 4 * r);
      float ev[16];
      float m = -3.4e38f;
      #pragma unroll
      for (int t = 0; t < 16; ++t) {
        int a = lane + 32 * t;
        float4 av4 = *(const float4*)(anch4 + 4 * a);
        float e = kv.x * av4.x + kv.y * av4.y + kv.z * av4.z + kv.w * av4.w;
        ev[t] = e; m = fmaxf(m, e);
      }
      m = wave_max(m);
      float s = 0.f;
      #pragma unroll
      for (int t = 0; t < 16; ++t) s += expf(ev[t] - m);
      s = wave_sum(s);
      float inv = 1.f / s;
      #pragma unroll
      for (int t = 0; t < 16; ++t)
        PkT[(lane + 32 * t) * 64 + r] = (__bf16)(expf(ev[t] - m) * inv);
    }
    __syncthreads();
    #pragma unroll
    for (int ks = 0; ks < 2; ++ks) {
      int kk = ks * 32;
      v16bf afr[3];
      #pragma unroll
      for (int mt = 0; mt < 3; ++mt) afr[mt] = load_a(pvbf + (mt * 16) * 64 + kk, 64);
      #pragma unroll
      for (int j = 0; j < 4; ++j) {
        int nt = wave * 4 + j;
        v16bf bfr = load_bt(PkT + (nt * 16) * 64 + kk, 64);
        #pragma unroll
        for (int mt = 0; mt < 3; ++mt)
          accz[mt][j] = __builtin_amdgcn_wmma_f32_16x16x32_bf16(
              false, afr[mt], false, bfr, (short)0, accz[mt][j], false, false);
      }
    }
    __syncthreads();
  }

  // ---- phase 2: spill z to LDS as bf16 A operand (row-major [M=48][K=512]) ----
  {
    int ncol = lane & 15;
    int mb = (lane < 16) ? 0 : 8;
    #pragma unroll
    for (int mt = 0; mt < 3; ++mt)
      #pragma unroll
      for (int j = 0; j < 4; ++j) {
        int nt = wave * 4 + j;
        #pragma unroll
        for (int r = 0; r < 8; ++r)
          zds[(mt * 16 + mb + r) * NA_ + nt * 16 + ncol] = (__bf16)accz[mt][j][r];
      }
  }
  __syncthreads();

  // ---- phase 3: out[48,2048] = z[48,512] x softmax(pq_e)[512,2048], chunked ----
  for (int ch = 0; ch < N_ / 128; ++ch) {
    int nc = ch * 128;
    for (int i = tid; i < CQ_ * 128; i += 256) {
      int c = i & 3, j = i >> 2;
      pqc4[i] = pqg[((size_t)b * CQ_ + c) * N_ + nc + j] * invq[b * CQ_ + c];
    }
    __syncthreads();
    // fill PQT[n=128][a=512]: 8 consecutive a per thread -> one 16B LDS store
    for (int ii = tid; ii < 128 * NA_ / 8; ii += 256) {
      int j = ii >> 6, a0 = (ii & 63) * 8;
      float4 qv = *(const float4*)(pqc4 + 4 * j);
      v8bf pkt;
      #pragma unroll
      for (int u = 0; u < 8; ++u) {
        int a = a0 + u;
        float4 av4 = *(const float4*)(anch4 + 4 * a);
        float e = av4.x * qv.x + av4.y * qv.y + av4.z * qv.z + av4.w * qv.w;
        pkt[u] = (__bf16)(expf(e - mq[a]) * sqv[a]);
      }
      *(v8bf*)(PQT + j * NA_ + a0) = pkt;
    }
    __syncthreads();
    v8f acco[3];
    #pragma unroll
    for (int mt = 0; mt < 3; ++mt) acco[mt] = zero8;
    #pragma unroll 4
    for (int ks = 0; ks < 16; ++ks) {
      int kk = ks * 32;
      v16bf bfr = load_bt(PQT + (wave * 16) * NA_ + kk, NA_);
      #pragma unroll
      for (int mt = 0; mt < 3; ++mt) {
        v16bf afr = load_a(zds + (mt * 16) * NA_ + kk, NA_);
        acco[mt] = __builtin_amdgcn_wmma_f32_16x16x32_bf16(
            false, afr, false, bfr, (short)0, acco[mt], false, false);
      }
    }
    int ncol = lane & 15;
    int mb = (lane < 16) ? 0 : 8;
    #pragma unroll
    for (int mt = 0; mt < 3; ++mt)
      #pragma unroll
      for (int r = 0; r < 8; ++r)
        attn[((size_t)b * C_ + mt * 16 + mb + r) * N_ + nc + wave * 16 + ncol] = acco[mt][r];
    __syncthreads();
  }
}

__global__ void __launch_bounds__(256) k_dwc(
    const float* __restrict__ pv, const float* __restrict__ w,
    float* __restrict__ dwc, float* bn1sum, float* bn1sq) {
  extern __shared__ __align__(16) char smem[];
  float* tile  = (float*)smem;        // 48*288 (halo 16 both sides)
  float* s_sum = tile + C_ * 288;
  float* s_sq  = s_sum + C_;
  int b = blockIdx.y, n0 = blockIdx.x * 256, tid = threadIdx.x;
  if (tid < C_) { s_sum[tid] = 0.f; s_sq[tid] = 0.f; }
  for (int i = tid; i < C_ * 288; i += 256) {
    int c = i / 288, j = i - c * 288;
    int n = n0 - 16 + j;
    tile[i] = (n >= 0 && n < N_) ? pv[((size_t)b * C_ + c) * N_ + n] : 0.f;
  }
  __syncthreads();
  for (int i = tid; i < C_ * 256; i += 256) {
    int c = i >> 8, j = i & 255;
    float acc = 0.f;
    #pragma unroll
    for (int t = 0; t < KW_; ++t) acc += w[c * KW_ + t] * tile[c * 288 + j + t];
    dwc[((size_t)b * C_ + c) * N_ + n0 + j] = acc;
    atomicAdd(&s_sum[c], acc);
    atomicAdd(&s_sq[c], acc * acc);
  }
  __syncthreads();
  if (tid < C_) { atomicAdd(&bn1sum[tid], s_sum[tid]); atomicAdd(&bn1sq[tid], s_sq[tid]); }
}

__global__ void k_bnstats(const float* sum, const float* sq, float* mu, float* rs,
                          float inv_count) {
  int c = threadIdx.x;
  if (c < C_) {
    float m = sum[c] * inv_count;
    float v = sq[c] * inv_count - m * m;
    mu[c] = m;
    rs[c] = rsqrtf(v + 1e-5f);
  }
}

__global__ void __launch_bounds__(256) k_combine(
    const float* __restrict__ attn, const float* __restrict__ dwc,
    const float* __restrict__ x, const float* __restrict__ av,
    const float* __restrict__ mu1, const float* __restrict__ rs1,
    const float* __restrict__ g1, const float* __restrict__ b1,
    float* __restrict__ out, float* bn2sum, float* bn2sq) {
  __shared__ float s_sum[C_], s_sq[C_];
  int b = blockIdx.y, n0 = blockIdx.x * 256, tid = threadIdx.x;
  if (tid < C_) { s_sum[tid] = 0.f; s_sq[tid] = 0.f; }
  __syncthreads();
  for (int i = tid; i < C_ * 256; i += 256) {
    int c = i >> 8, j = i & 255;
    size_t idx = ((size_t)b * C_ + c) * N_ + n0 + j;
    float d = (dwc[idx] - mu1[c]) * rs1[c] * g1[c] + b1[c];
    d = eluf(d);
    float pre = attn[idx] + d + av[b * C_ + c] * x[idx];
    out[idx] = pre;
    atomicAdd(&s_sum[c], pre);
    atomicAdd(&s_sq[c], pre * pre);
  }
  __syncthreads();
  if (tid < C_) { atomicAdd(&bn2sum[tid], s_sum[tid]); atomicAdd(&bn2sq[tid], s_sq[tid]); }
}

__global__ void k_bn_apply(float* __restrict__ out, const float* __restrict__ mu,
                           const float* __restrict__ rs, const float* __restrict__ g,
                           const float* __restrict__ bb) {
  int b = blockIdx.y, n0 = blockIdx.x * 256, tid = threadIdx.x;
  for (int i = tid; i < C_ * 256; i += 256) {
    int c = i >> 8;
    size_t idx = ((size_t)b * C_ + c) * N_ + n0 + (i & 255);
    out[idx] = (out[idx] - mu[c]) * rs[c] * g[c] + bb[c];
  }
}

// ---------------- launcher ----------------
extern "C" void kernel_launch(void* const* d_in, const int* in_sizes, int n_in,
                              void* d_out, int out_size, void* d_ws, size_t ws_size,
                              hipStream_t stream) {
  (void)in_sizes; (void)n_in; (void)out_size; (void)ws_size;
  const float* x     = (const float*)d_in[0];
  const float* ca1_w = (const float*)d_in[1];
  const float* ca1_b = (const float*)d_in[2];
  const float* ca2_w = (const float*)d_in[3];
  const float* ca2_b = (const float*)d_in[4];
  const float* q_w   = (const float*)d_in[5];
  const float* q_b   = (const float*)d_in[6];
  const float* k_w   = (const float*)d_in[7];
  const float* k_b   = (const float*)d_in[8];
  const float* v_w   = (const float*)d_in[9];
  const float* v_b   = (const float*)d_in[10];
  const float* a_w   = (const float*)d_in[11];
  const float* a_b   = (const float*)d_in[12];
  const float* dwc_w = (const float*)d_in[13];
  const float* dwc_g = (const float*)d_in[14];
  const float* dwc_b = (const float*)d_in[15];
  const float* bn_g  = (const float*)d_in[16];
  const float* bn_b  = (const float*)d_in[17];
  float* ws  = (float*)d_ws;
  float* out = (float*)d_out;

  hipFuncSetAttribute((const void*)k_proj, hipFuncAttributeMaxDynamicSharedMemorySize, SMEM_PROJ);
  hipFuncSetAttribute((const void*)k_attention, hipFuncAttributeMaxDynamicSharedMemorySize, SMEM_ATTN);
  hipFuncSetAttribute((const void*)k_dwc, hipFuncAttributeMaxDynamicSharedMemorySize, SMEM_DWC);

  k_zero<<<1, 256, 0, stream>>>(ws);
  k_se<<<B_, 256, 0, stream>>>(x, ca1_w, ca1_b, ca2_w, ca2_b, ws + OFF_AV);
  k_proj<<<dim3(N_ / 256, B_), 256, SMEM_PROJ, stream>>>(
      x, ws + OFF_AV, k_w, k_b, q_w, q_b, v_w, v_b, a_w, a_b,
      ws + OFF_PK, ws + OFF_PQ, ws + OFF_PV, ws + OFF_ANC,
      ws + OFF_SQK, ws + OFF_SQQ, ws + OFF_SQA);
  k_norms<<<1, 128, 0, stream>>>(ws);
  k_attention<<<B_, 256, SMEM_ATTN, stream>>>(
      ws + OFF_PK, ws + OFF_PQ, ws + OFF_ANC, ws + OFF_PV,
      ws + OFF_INVK, ws + OFF_INVQ, ws + OFF_INVA, ws + OFF_ATTN);
  k_dwc<<<dim3(N_ / 256, B_), 256, SMEM_DWC, stream>>>(
      ws + OFF_PV, dwc_w, ws + OFF_DWC, ws + OFF_BN1SUM, ws + OFF_BN1SQ);
  k_bnstats<<<1, 64, 0, stream>>>(ws + OFF_BN1SUM, ws + OFF_BN1SQ,
                                  ws + OFF_BN1MU, ws + OFF_BN1RS, 1.f / (B_ * N_));
  k_combine<<<dim3(N_ / 256, B_), 256, 0, stream>>>(
      ws + OFF_ATTN, ws + OFF_DWC, x, ws + OFF_AV,
      ws + OFF_BN1MU, ws + OFF_BN1RS, dwc_g, dwc_b, out,
      ws + OFF_BN2SUM, ws + OFF_BN2SQ);
  k_bnstats<<<1, 64, 0, stream>>>(ws + OFF_BN2SUM, ws + OFF_BN2SQ,
                                  ws + OFF_BN2MU, ws + OFF_BN2RS, 1.f / (B_ * N_));
  k_bn_apply<<<dim3(N_ / 256, B_), 256, 0, stream>>>(out, ws + OFF_BN2MU, ws + OFF_BN2RS,
                                                     bn_g, bn_b);
}
